// CausalSelfAttention_11175504904425
// MI455X (gfx1250) — matrix operand order repaired and checked by reading, not compiled, not run on hardware
//
#include <hip/hip_runtime.h>
#include <hip/hip_bf16.h>

// ---------------------------------------------------------------------------
// Types for CDNA5 WMMA
// ---------------------------------------------------------------------------
typedef __attribute__((ext_vector_type(16))) __bf16 v16bf;
typedef __attribute__((ext_vector_type(8)))  float  v8f;
typedef int v4i_ __attribute__((vector_size(16)));   // matches builtin param type

#define AS1 __attribute__((address_space(1)))
#define AS3 __attribute__((address_space(3)))

union FragU { uint4 q[2]; v16bf v; };

__device__ __forceinline__ v16bf mkfrag(uint4 a, uint4 b) {
    FragU f; f.q[0] = a; f.q[1] = b; return f.v;
}

__device__ __forceinline__ v8f wmma_bf16(v16bf a, v16bf b, v8f c) {
    // D = A(16x32 bf16) * B(32x16 bf16) + C(16x16 f32)
    return __builtin_amdgcn_wmma_f32_16x16x32_bf16(
        false, a, false, b, (short)0, c, false, false);
}

__device__ __forceinline__ unsigned short f2bf(float f) {
    unsigned int u = __float_as_uint(f);
    u = (u + 0x7FFFu + ((u >> 16) & 1u)) >> 16;   // round-to-nearest-even
    return (unsigned short)u;
}
__device__ __forceinline__ float bf2f(unsigned short h) {
    return __uint_as_float(((unsigned int)h) << 16);
}

// ---------------------------------------------------------------------------
// Async global -> LDS copy (CDNA5 path, ASYNCcnt-tracked, no VGPR round-trip)
// ---------------------------------------------------------------------------
#if defined(__has_builtin)
#if __has_builtin(__builtin_amdgcn_global_load_async_to_lds_b128)
#define HAS_ASYNC_B128 1
#endif
#if __has_builtin(__builtin_amdgcn_s_wait_asynccnt)
#define HAS_WAIT_ASYNC 1
#endif
#endif

__device__ __forceinline__ void async_cp16(void* lds, const void* g) {
#ifdef HAS_ASYNC_B128
    __builtin_amdgcn_global_load_async_to_lds_b128(
        (AS1 v4i_*)g, (AS3 v4i_*)lds, 0, 0);
#else
    unsigned loff = (unsigned)(unsigned long long)lds;
    asm volatile("global_load_async_to_lds_b128 %0, %1, off"
                 :: "v"(loff), "v"(g) : "memory");
#endif
}

#ifdef HAS_WAIT_ASYNC
#define WAIT_ASYNC(n) __builtin_amdgcn_s_wait_asynccnt(n)
#else
#define WAIT_ASYNC(n) asm volatile("s_wait_asynccnt %0" :: "n"(n) : "memory")
#endif

// Problem constants
#define BB   2
#define TT   2048
#define CC   1024
#define HH   16
#define HD   64

// ---------------------------------------------------------------------------
// f32 -> bf16 conversion (n is a multiple of 4)
// ---------------------------------------------------------------------------
__global__ __launch_bounds__(256) void k_cvt_bf16(
    const float* __restrict__ in, unsigned short* __restrict__ out, int n4) {
    int i = blockIdx.x * 256 + threadIdx.x;
    if (i < n4) {
        float4 v = ((const float4*)in)[i];
        uint2 r;
        r.x = (unsigned)f2bf(v.x) | ((unsigned)f2bf(v.y) << 16);
        r.y = (unsigned)f2bf(v.z) | ((unsigned)f2bf(v.w) << 16);
        ((uint2*)out)[i] = r;
    }
}

// ---------------------------------------------------------------------------
// GEMM: C[M,N] = A[M,K] * Bw[N,K]^T  (bf16 in, f32 accum, f32 or bf16 out)
// block = 256 threads (8 waves), tile 128x128, K-step 32.
// Async double-buffered LDS staging: copy tile k+1 while WMMA consumes tile k.
// ---------------------------------------------------------------------------
template <bool F32OUT>
__global__ __launch_bounds__(256) void k_gemm_bf16(
    const unsigned short* __restrict__ A,
    const unsigned short* __restrict__ Bw,
    float* __restrict__ Cf,
    unsigned short* __restrict__ Cb,
    int M, int N, int K) {
    __shared__ unsigned short As[2][128][40];   // padded rows (80B = 5*16B)
    __shared__ unsigned short Bs[2][128][40];

    const int tid  = threadIdx.x;
    const int wave = tid >> 5;
    const int lane = tid & 31;
    const int hi   = lane >> 4;        // half-wave select
    const int lo   = lane & 15;
    const int bm   = blockIdx.y * 128;
    const int bn   = blockIdx.x * 128;
    const int wm   = (wave >> 1) * 32; // wave M offset (4 rows of waves)
    const int wn   = (wave & 1) * 64;  // wave N offset (2 cols of waves)

    // per-thread copy assignment: 2 chunks of A + 2 of B per tile
    const int cr0 = tid >> 2;                 // 0..63
    const int cr1 = (tid + 256) >> 2;         // 64..127
    const int ccc = (tid & 3) * 8;            // 0,8,16,24

    auto issue_tile = [&](int kt, int buf) {
        const int k0 = kt << 5;
        async_cp16(&As[buf][cr0][ccc], A  + (size_t)(bm + cr0) * K + k0 + ccc);
        async_cp16(&As[buf][cr1][ccc], A  + (size_t)(bm + cr1) * K + k0 + ccc);
        async_cp16(&Bs[buf][cr0][ccc], Bw + (size_t)(bn + cr0) * K + k0 + ccc);
        async_cp16(&Bs[buf][cr1][ccc], Bw + (size_t)(bn + cr1) * K + k0 + ccc);
    };

    v8f acc[2][4];
    #pragma unroll
    for (int i = 0; i < 2; ++i)
        #pragma unroll
        for (int j = 0; j < 4; ++j) acc[i][j] = (v8f)(0.0f);

    const int steps = K >> 5;
    issue_tile(0, 0);

    for (int kt = 0; kt < steps; ++kt) {
        const int buf = kt & 1;
        __syncthreads();                       // all waves done reading buf^1
        if (kt + 1 < steps) {
            issue_tile(kt + 1, buf ^ 1);
            WAIT_ASYNC(4);                     // own tile-kt copies complete
        } else {
            WAIT_ASYNC(0);
        }
        __syncthreads();                       // tile kt visible to all waves

        v16bf af[2], bfv[4];
        #pragma unroll
        for (int mt = 0; mt < 2; ++mt) {
            // A-frag: lane<16 -> k 0..7 & 16..23 ; lane>=16 -> k 8..15 & 24..31
            const unsigned short* p = &As[buf][wm + mt * 16 + lo][hi * 8];
            af[mt] = mkfrag(*(const uint4*)p, *(const uint4*)(p + 16));
        }
        #pragma unroll
        for (int nt = 0; nt < 4; ++nt) {
            // B-frag: lane<16 -> k 0..7 & 16..23 ; lane>=16 -> k 8..15 & 24..31
            const unsigned short* p = &Bs[buf][wn + nt * 16 + lo][hi * 8];
            bfv[nt] = mkfrag(*(const uint4*)p, *(const uint4*)(p + 16));
        }
        #pragma unroll
        for (int mt = 0; mt < 2; ++mt)
            #pragma unroll
            for (int nt = 0; nt < 4; ++nt)
                acc[mt][nt] = wmma_bf16(af[mt], bfv[nt], acc[mt][nt]);
    }

    // Epilogue: C frag layout -> vgpr r, lane<16: row r, lane>=16: row r+8
    #pragma unroll
    for (int mt = 0; mt < 2; ++mt)
        #pragma unroll
        for (int nt = 0; nt < 4; ++nt)
            #pragma unroll
            for (int r = 0; r < 8; ++r) {
                int row = bm + wm + mt * 16 + hi * 8 + r;
                int col = bn + wn + nt * 16 + lo;
                float v = acc[mt][nt][r];
                if constexpr (F32OUT) Cf[(size_t)row * N + col] = v;
                else                  Cb[(size_t)row * N + col] = f2bf(v);
            }
}

// ---------------------------------------------------------------------------
// RMSNorm + RoPE + layout change.
// One wave per (b,t,h): q,k normalized+rotated -> (b,h,t,d) bf16;
// v -> transposed (b,h,d,t) bf16 so P@V B-frags are contiguous in the key dim.
// ---------------------------------------------------------------------------
__global__ __launch_bounds__(256) void k_prep(
    const unsigned short* __restrict__ qkvb,   // (B*T, 3C) bf16
    const float* __restrict__ fc,              // (T, 32)
    const float* __restrict__ fs,              // (T, 32)
    const float* __restrict__ qw,              // (64)
    const float* __restrict__ kw,              // (64)
    unsigned short* __restrict__ qh,           // (B,H,T,64)
    unsigned short* __restrict__ kh,           // (B,H,T,64)
    unsigned short* __restrict__ vT) {         // (B,H,64,T)
    const int tid  = threadIdx.x;
    const int wave = tid >> 5;
    const int lane = tid & 31;
    const int idx  = blockIdx.x * 8 + wave;    // b*T*H + t*H + h
    const int h = idx & (HH - 1);
    const int t = (idx >> 4) & (TT - 1);
    const int b = idx >> 15;

    const unsigned short* row = qkvb + (size_t)(b * TT + t) * (3 * CC);
    const float c  = fc[t * 32 + lane];
    const float s  = fs[t * 32 + lane];
    const int   bh = b * HH + h;

    #pragma unroll
    for (int which = 0; which < 2; ++which) {   // 0 = q, 1 = k
        const int coff = which * CC + h * HD + 2 * lane;
        unsigned int u = *(const unsigned int*)(row + coff);
        float x0 = bf2f((unsigned short)(u & 0xFFFFu));
        float x1 = bf2f((unsigned short)(u >> 16));
        float ss = x0 * x0 + x1 * x1;
        #pragma unroll
        for (int m = 1; m < 32; m <<= 1) ss += __shfl_xor(ss, m, 32);
        float rms = rsqrtf(ss * (1.0f / HD) + 1e-6f);
        const float* w = which ? kw : qw;
        float n0 = x0 * rms * w[2 * lane];
        float n1 = x1 * rms * w[2 * lane + 1];
        float r0 = n0 * c - n1 * s;
        float r1 = n0 * s + n1 * c;
        unsigned int pk = (unsigned)f2bf(r0) | ((unsigned)f2bf(r1) << 16);
        unsigned short* dst = which ? kh : qh;
        *(unsigned int*)(dst + ((size_t)bh * TT + t) * HD + 2 * lane) = pk;
    }
    // V: copy (transposed to d-major)
    unsigned int uv = *(const unsigned int*)(row + 2 * CC + h * HD + 2 * lane);
    vT[((size_t)bh * HD + 2 * lane    ) * TT + t] = (unsigned short)(uv & 0xFFFFu);
    vT[((size_t)bh * HD + 2 * lane + 1) * TT + t] = (unsigned short)(uv >> 16);
}

// ---------------------------------------------------------------------------
// Flash attention: one block per (b,h, 128 query rows); 8 waves x 16 rows.
// Online softmax in f32; S = Q K^T and O += P V on WMMA bf16.
// K/V tiles streamed with async-to-LDS copies, double-buffered.
// ---------------------------------------------------------------------------
__global__ __launch_bounds__(256) void k_attn(
    const unsigned short* __restrict__ qh,   // (B,H,T,64)
    const unsigned short* __restrict__ kh,   // (B,H,T,64)
    const unsigned short* __restrict__ vT,   // (B,H,64,T)
    unsigned short* __restrict__ yb) {       // (B,T,C) bf16
    __shared__ unsigned short Kt[2][64][72]; // key tile, row = key, col = d
    __shared__ unsigned short Vt[2][64][72]; // value tile, row = d, col = key
    __shared__ unsigned short Pt[8][16][64]; // per-wave P scratch

    const int tid  = threadIdx.x;
    const int wave = tid >> 5;
    const int lane = tid & 31;
    const int hi   = lane >> 4;
    const int lo   = lane & 15;
    const int bh   = blockIdx.y;             // b*H + h
    const int b    = bh >> 4;
    const int h    = bh & 15;
    const int t0   = blockIdx.x * 128;

    const unsigned short* qbase = qh + (size_t)bh * TT * HD;
    const unsigned short* kbase = kh + (size_t)bh * TT * HD;
    const unsigned short* vbase = vT + (size_t)bh * HD * TT;

    // per-thread copy assignment: rows 0..63, 8 chunks of 8 shorts
    const int lr0 = tid >> 3;                // 0..31
    const int lr1 = (tid + 256) >> 3;        // 32..63
    const int lcc = (tid & 7) * 8;           // 0..56

    auto issue_tile = [&](int j0, int buf) {
        async_cp16(&Kt[buf][lr0][lcc], kbase + (size_t)(j0 + lr0) * HD + lcc);
        async_cp16(&Kt[buf][lr1][lcc], kbase + (size_t)(j0 + lr1) * HD + lcc);
        async_cp16(&Vt[buf][lr0][lcc], vbase + (size_t)lr0 * TT + j0 + lcc);
        async_cp16(&Vt[buf][lr1][lcc], vbase + (size_t)lr1 * TT + j0 + lcc);
    };

    // Q fragments (kept in registers for the whole pass)
    v16bf aq[2];
    const int qrow = t0 + wave * 16 + lo;
    #pragma unroll
    for (int ks = 0; ks < 2; ++ks) {
        const unsigned short* p = qbase + (size_t)qrow * HD + ks * 32 + hi * 8;
        aq[ks] = mkfrag(*(const uint4*)p, *(const uint4*)(p + 16));
    }

    v8f o[4];
    #pragma unroll
    for (int nt = 0; nt < 4; ++nt) o[nt] = (v8f)(0.0f);
    float mrow[8], lrow[8];
    #pragma unroll
    for (int r = 0; r < 8; ++r) { mrow[r] = -3.0e38f; lrow[r] = 0.0f; }

    const int njt = (t0 + 128) >> 6;         // causal: keys < t0+128 suffice
    issue_tile(0, 0);

    for (int jt = 0; jt < njt; ++jt) {
        const int j0  = jt << 6;
        const int buf = jt & 1;
        __syncthreads();                     // all waves done reading buf^1
        if (jt + 1 < njt) {
            issue_tile(j0 + 64, buf ^ 1);
            WAIT_ASYNC(4);                   // own tile-jt copies complete
        } else {
            WAIT_ASYNC(0);
        }
        __syncthreads();                     // tile jt visible to all waves

        // ---- S = Q K^T (16x64 per wave) ----
        v8f sc[4];
        #pragma unroll
        for (int nt = 0; nt < 4; ++nt) sc[nt] = (v8f)(0.0f);
        #pragma unroll
        for (int ks = 0; ks < 2; ++ks)
            #pragma unroll
            for (int nt = 0; nt < 4; ++nt) {
                const unsigned short* p = &Kt[buf][nt * 16 + lo][ks * 32 + hi * 8];
                v16bf bk = mkfrag(*(const uint4*)p, *(const uint4*)(p + 16));
                sc[nt] = wmma_bf16(aq[ks], bk, sc[nt]);
            }

        // ---- scale + causal mask + online softmax ----
        #pragma unroll
        for (int r = 0; r < 8; ++r) {
            const int mg = t0 + wave * 16 + hi * 8 + r;   // global query row
            float vmax = -3.0e38f;
            #pragma unroll
            for (int nt = 0; nt < 4; ++nt) {
                float sv = sc[nt][r] * 0.125f;            // 1/sqrt(64)
                int ng = j0 + nt * 16 + lo;
                sv = (ng > mg) ? -3.0e38f : sv;
                sc[nt][r] = sv;
                vmax = fmaxf(vmax, sv);
            }
            #pragma unroll
            for (int m = 1; m < 16; m <<= 1)
                vmax = fmaxf(vmax, __shfl_xor(vmax, m, 32));
            float mn = fmaxf(mrow[r], vmax);
            float cf = __expf(mrow[r] - mn);
            mrow[r] = mn;
            float ls = 0.0f;
            #pragma unroll
            for (int nt = 0; nt < 4; ++nt) {
                float p = __expf(sc[nt][r] - mn);
                sc[nt][r] = p;
                ls += p;
            }
            #pragma unroll
            for (int m = 1; m < 16; m <<= 1) ls += __shfl_xor(ls, m, 32);
            lrow[r] = lrow[r] * cf + ls;
            #pragma unroll
            for (int nt = 0; nt < 4; ++nt) o[nt][r] *= cf;
            // write P (bf16) to per-wave LDS for C-layout -> A-layout reshuffle
            #pragma unroll
            for (int nt = 0; nt < 4; ++nt)
                Pt[wave][hi * 8 + r][nt * 16 + lo] = f2bf(sc[nt][r]);
        }

        // ---- O += P V ----
        #pragma unroll
        for (int ks = 0; ks < 2; ++ks) {
            const unsigned short* pp = &Pt[wave][lo][ks * 32 + hi * 8];
            v16bf ap = mkfrag(*(const uint4*)pp, *(const uint4*)(pp + 16));
            #pragma unroll
            for (int nt = 0; nt < 4; ++nt) {
                const unsigned short* vp = &Vt[buf][nt * 16 + lo][ks * 32 + hi * 8];
                v16bf bv = mkfrag(*(const uint4*)vp, *(const uint4*)(vp + 16));
                o[nt] = wmma_bf16(ap, bv, o[nt]);
            }
        }
    }

    // ---- epilogue: O / l, write bf16 y in (B,T,C) ----
    #pragma unroll
    for (int nt = 0; nt < 4; ++nt)
        #pragma unroll
        for (int r = 0; r < 8; ++r) {
            float val = o[nt][r] / lrow[r];
            int trow = t0 + wave * 16 + hi * 8 + r;
            int col  = h * HD + nt * 16 + lo;
            yb[(size_t)(b * TT + trow) * CC + col] = f2bf(val);
        }
}

// ---------------------------------------------------------------------------
// Host-side launcher
// ---------------------------------------------------------------------------
extern "C" void kernel_launch(void* const* d_in, const int* in_sizes, int n_in,
                              void* d_out, int out_size, void* d_ws, size_t ws_size,
                              hipStream_t stream) {
    (void)in_sizes; (void)n_in; (void)out_size; (void)ws_size;
    const float* x      = (const float*)d_in[0];
    const float* fc     = (const float*)d_in[1];
    const float* fs     = (const float*)d_in[2];
    const float* w_attn = (const float*)d_in[3];
    const float* w_proj = (const float*)d_in[4];
    const float* qw     = (const float*)d_in[5];
    const float* kw     = (const float*)d_in[6];
    float* out = (float*)d_out;

    char* ws = (char*)d_ws;
    const size_t MB = 1ull << 20;
    unsigned short* xb   = (unsigned short*)(ws);             //  8 MB (B*T,C)
    unsigned short* wab  = (unsigned short*)(ws +  8 * MB);   //  6 MB (3C,C)
    unsigned short* wpb  = (unsigned short*)(ws + 14 * MB);   //  2 MB (C,C)
    unsigned short* qkvb = (unsigned short*)(ws + 16 * MB);   // 24 MB (B*T,3C)
    unsigned short* qhb  = (unsigned short*)(ws + 40 * MB);   //  8 MB (B,H,T,64)
    unsigned short* khb  = (unsigned short*)(ws + 48 * MB);   //  8 MB
    unsigned short* vTb  = (unsigned short*)(ws + 56 * MB);   //  8 MB (B,H,64,T)
    unsigned short* ybb  = (unsigned short*)(ws + 64 * MB);   //  8 MB (B,T,C)

    const int M = BB * TT;      // 4096
    // 1) f32 -> bf16 conversions
    k_cvt_bf16<<<(M * CC / 4 + 255) / 256, 256, 0, stream>>>(x, xb, M * CC / 4);
    k_cvt_bf16<<<(3 * CC * CC / 4 + 255) / 256, 256, 0, stream>>>(w_attn, wab, 3 * CC * CC / 4);
    k_cvt_bf16<<<(CC * CC / 4 + 255) / 256, 256, 0, stream>>>(w_proj, wpb, CC * CC / 4);

    // 2) qkv = x @ w_attn^T  (4096 x 3072 x 1024), bf16 out
    k_gemm_bf16<false><<<dim3(3 * CC / 128, M / 128), 256, 0, stream>>>(
        xb, wab, nullptr, qkvb, M, 3 * CC, CC);

    // 3) RMSNorm + RoPE + layout
    k_prep<<<BB * TT * HH / 8, 256, 0, stream>>>(qkvb, fc, fs, qw, kw, qhb, khb, vTb);

    // 4) causal flash attention
    k_attn<<<dim3(TT / 128, BB * HH), 256, 0, stream>>>(qhb, khb, vTb, ybb);

    // 5) out = y @ w_proj^T  (4096 x 1024 x 1024), f32 out
    k_gemm_bf16<true><<<dim3(CC / 128, M / 128), 256, 0, stream>>>(
        ybb, wpb, out, nullptr, M, CC, CC);
}
